// MemoryEfficientCLIPLoss_31258771980808
// MI455X (gfx1250) — matrix operand order repaired and checked by reading, not compiled, not run on hardware
//
#include <hip/hip_runtime.h>
#include <stdint.h>

// ---------------------------------------------------------------------------
// Memory-efficient CLIP loss for MI455X (gfx1250, wave32, WMMA, async-LDS).
//   B = 8192 rows, D = 512 features, fp32 inputs, scalar fp32 output.
//   Phase 0: L2-normalize rows, convert to bf16, compute diagonal Sv.
//   Phase 1: tiled bf16 WMMA over the 8192x8192 similarity matrix with
//            double-buffered global_load_async_to_lds_b128 staging,
//            E = exp2(S*it - it), accumulate row/col sums (flash style).
//   Phase 2: loss = -mean(2*Sv - log(rowSum) - log(colSum)) / 2.
// ---------------------------------------------------------------------------

#define B_ROWS   8192
#define D_DIM    512
#define INV_TEMP 20.609929155556620f   // log2(e) / 0.07

typedef __attribute__((ext_vector_type(16))) __bf16 v16bf;
typedef __attribute__((ext_vector_type(8)))  __bf16 v8bf;
typedef __attribute__((ext_vector_type(8)))  float  v8f;

// ---------------------------------------------------------------------------
// Phase 0: one 128-thread block per row.
// ---------------------------------------------------------------------------
__global__ void __launch_bounds__(128)
clip_normalize_kernel(const float* __restrict__ X, const float* __restrict__ Y,
                      __bf16* __restrict__ Xnb, __bf16* __restrict__ Ynb,
                      float* __restrict__ Sv, float* __restrict__ rowSum,
                      float* __restrict__ colSum) {
    const int row  = blockIdx.x;
    const int tid  = threadIdx.x;            // 0..127
    const int lane = tid & 31;
    const int wave = tid >> 5;               // 0..3

    const float4 x4 = ((const float4*)(X + (size_t)row * D_DIM))[tid];
    const float4 y4 = ((const float4*)(Y + (size_t)row * D_DIM))[tid];

    float sxx = x4.x * x4.x + x4.y * x4.y + x4.z * x4.z + x4.w * x4.w;
    float syy = y4.x * y4.x + y4.y * y4.y + y4.z * y4.z + y4.w * y4.w;
    float sxy = x4.x * y4.x + x4.y * y4.y + x4.z * y4.z + x4.w * y4.w;

#pragma unroll
    for (int off = 16; off >= 1; off >>= 1) {
        sxx += __shfl_xor(sxx, off, 32);
        syy += __shfl_xor(syy, off, 32);
        sxy += __shfl_xor(sxy, off, 32);
    }

    __shared__ float red[3][4];
    if (lane == 0) { red[0][wave] = sxx; red[1][wave] = syy; red[2][wave] = sxy; }
    __syncthreads();

    const float tx  = red[0][0] + red[0][1] + red[0][2] + red[0][3];
    const float ty  = red[1][0] + red[1][1] + red[1][2] + red[1][3];
    const float txy = red[2][0] + red[2][1] + red[2][2] + red[2][3];
    const float rx = rsqrtf(tx);
    const float ry = rsqrtf(ty);

    if (tid == 0) {
        Sv[row]     = txy * rx * ry * INV_TEMP - INV_TEMP;
        rowSum[row] = 0.0f;
        colSum[row] = 0.0f;
    }

    __bf16* xo = Xnb + (size_t)row * D_DIM + tid * 4;
    __bf16* yo = Ynb + (size_t)row * D_DIM + tid * 4;
    xo[0] = (__bf16)(x4.x * rx); xo[1] = (__bf16)(x4.y * rx);
    xo[2] = (__bf16)(x4.z * rx); xo[3] = (__bf16)(x4.w * rx);
    yo[0] = (__bf16)(y4.x * ry); yo[1] = (__bf16)(y4.y * ry);
    yo[2] = (__bf16)(y4.z * ry); yo[3] = (__bf16)(y4.w * ry);
}

// ---------------------------------------------------------------------------
// Phase 1: 64x64 grid of 256-thread workgroups; each owns a 128x128 tile.
// K panels (128x32 bf16 for A and B, 8 KB each) are staged into LDS with
// double-buffered async loads; fragments are read back with ds_load_b128.
//
// Dynamic LDS layout (byte offsets):
//   [    0,  8192)  A panel, buffer 0   (128 rows x 32 K, row stride 64 B)
//   [ 8192, 16384)  B panel, buffer 0
//   [16384, 24576)  A panel, buffer 1
//   [24576, 32768)  B panel, buffer 1
//   [32768, 33280)  rowAcc (128 f32)
//   [33280, 33792)  colAcc (128 f32)
// ---------------------------------------------------------------------------
#define ABUF(p)     ((unsigned)(p) * 16384u)
#define BBUF(p)     ((unsigned)(p) * 16384u + 8192u)
#define ROWACC_OFF  32768u
#define COLACC_OFF  33280u
#define LDS_BYTES   33792u

extern __shared__ char smem[];

__device__ __forceinline__ void async_ld16(unsigned lds_off, const __bf16* g) {
    asm volatile("global_load_async_to_lds_b128 %0, %1, off"
                 :: "v"(lds_off), "v"((unsigned long long)(uintptr_t)g)
                 : "memory");
}

__global__ void __launch_bounds__(256, 2)
clip_tile_kernel(const __bf16* __restrict__ A, const __bf16* __restrict__ Bm,
                 float* __restrict__ rowSum, float* __restrict__ colSum) {
    const int tid = threadIdx.x;
    float* rowAcc = (float*)(smem + ROWACC_OFF);
    float* colAcc = (float*)(smem + COLACC_OFF);
    if (tid < 128) rowAcc[tid] = 0.0f; else colAcc[tid - 128] = 0.0f;

    const int lane = tid & 31;
    const int wave = tid >> 5;
    const int wRow = wave >> 1;     // 0..3  -> 32-row band
    const int wCol = wave & 1;      // 0..1  -> 64-col band
    const int l15  = lane & 15;
    const int hi   = lane >> 4;

    const int rowBase = blockIdx.y * 128;
    const int colBase = blockIdx.x * 128;

    // --- staging map: thread t moves rows (t>>2) and (t>>2)+64, 16B chunk t&3
    const int srow = tid >> 2;                 // 0..63
    const int soff = tid & 3;                  // 16-byte chunk within 64B row
    const __bf16* gA0 = A  + (size_t)(rowBase + srow) * D_DIM + soff * 8;
    const __bf16* gA1 = gA0 + (size_t)64 * D_DIM;
    const __bf16* gB0 = Bm + (size_t)(colBase + srow) * D_DIM + soff * 8;
    const __bf16* gB1 = gB0 + (size_t)64 * D_DIM;
    const unsigned lds0 = (unsigned)(srow * 64 + soff * 16);
    const unsigned lds1 = lds0 + 64u * 64u;

    // prologue: stage K-step 0 into buffer 0
    async_ld16(ABUF(0) + lds0, gA0);
    async_ld16(ABUF(0) + lds1, gA1);
    async_ld16(BBUF(0) + lds0, gB0);
    async_ld16(BBUF(0) + lds1, gB1);

    // fp32 accumulators: 2 row-tiles x 4 col-tiles of 16x16
    v8f acc[2][4];
#pragma unroll
    for (int r = 0; r < 2; ++r)
#pragma unroll
        for (int c = 0; c < 4; ++c)
#pragma unroll
            for (int v = 0; v < 8; ++v) acc[r][c][v] = 0.0f;

    // fragment base byte offsets inside a panel
    //   A: lane -> row wRow*32 + r*16 + l15, K = hi*8 + {0..7} and {16..23}
    //   B: lane -> col wCol*64 + c*16 + l15, K = hi*8 + {0..7} and {16..23}
    const unsigned aOffBase = (unsigned)((wRow * 32 + l15) * 64 + hi * 16);
    const unsigned bOffBase = (unsigned)((wCol * 64 + l15) * 64 + hi * 16);

    for (int k = 0; k < D_DIM; k += 32) {
        const int p = (k >> 5) & 1;

        if (k + 32 < D_DIM) {
            // stage next K panel into the other buffer (safe: all reads of that
            // buffer finished at the trailing barrier of the previous iter)
            const int kn = k + 32;
            async_ld16(ABUF(p ^ 1) + lds0, gA0 + kn);
            async_ld16(ABUF(p ^ 1) + lds1, gA1 + kn);
            async_ld16(BBUF(p ^ 1) + lds0, gB0 + kn);
            async_ld16(BBUF(p ^ 1) + lds1, gB1 + kn);
            asm volatile("s_wait_asynccnt 0x4" ::: "memory");   // current stage done
        } else {
            asm volatile("s_wait_asynccnt 0x0" ::: "memory");
        }
        __syncthreads();   // all waves' async writes for this panel visible

        const char* pa = smem + ABUF(p);
        const char* pb = smem + BBUF(p);

        v16bf af[2], bfr[4];
#pragma unroll
        for (int r = 0; r < 2; ++r) {
            const unsigned base = aOffBase + (unsigned)r * 1024u;
            v8bf lo = *(const v8bf*)(pa + base);
            v8bf hh = *(const v8bf*)(pa + base + 32);
#pragma unroll
            for (int i = 0; i < 8; ++i) { af[r][i] = lo[i]; af[r][i + 8] = hh[i]; }
        }
#pragma unroll
        for (int c = 0; c < 4; ++c) {
            const unsigned base = bOffBase + (unsigned)c * 1024u;
            v8bf lo = *(const v8bf*)(pb + base);
            v8bf hh = *(const v8bf*)(pb + base + 32);
#pragma unroll
            for (int i = 0; i < 8; ++i) { bfr[c][i] = lo[i]; bfr[c][i + 8] = hh[i]; }
        }

#pragma unroll
        for (int r = 0; r < 2; ++r)
#pragma unroll
            for (int c = 0; c < 4; ++c)
                acc[r][c] = __builtin_amdgcn_wmma_f32_16x16x32_bf16(
                    false, af[r], false, bfr[c], (short)0, acc[r][c], false, false);

        __syncthreads();   // all waves done reading panel p before it is reused
    }

    // E = exp2(S * it - it)
#pragma unroll
    for (int r = 0; r < 2; ++r)
#pragma unroll
        for (int c = 0; c < 4; ++c)
#pragma unroll
            for (int v = 0; v < 8; ++v)
                acc[r][c][v] = __builtin_amdgcn_exp2f(acc[r][c][v] * INV_TEMP - INV_TEMP);

    // Column partial sums: lane's column is wCol*64 + c*16 + l15 for both halves.
#pragma unroll
    for (int c = 0; c < 4; ++c) {
        float cp = 0.0f;
#pragma unroll
        for (int r = 0; r < 2; ++r)
#pragma unroll
            for (int v = 0; v < 8; ++v) cp += acc[r][c][v];
        atomicAdd(&colAcc[wCol * 64 + c * 16 + l15], cp);
    }

    // Row partial sums: sum the wave's 4 col-tiles in registers, 16-lane
    // butterfly (stays within each half-wave), then one LDS atomic.
#pragma unroll
    for (int r = 0; r < 2; ++r)
#pragma unroll
        for (int v = 0; v < 8; ++v) {
            float rp = 0.0f;
#pragma unroll
            for (int c = 0; c < 4; ++c) rp += acc[r][c][v];
#pragma unroll
            for (int off = 8; off >= 1; off >>= 1) rp += __shfl_xor(rp, off, 32);
            if (l15 == 0)
                atomicAdd(&rowAcc[wRow * 32 + r * 16 + hi * 8 + v], rp);
        }

    __syncthreads();
    if (tid < 128) atomicAdd(&rowSum[rowBase + tid], rowAcc[tid]);
    else           atomicAdd(&colSum[colBase + tid - 128], colAcc[tid - 128]);
}

// ---------------------------------------------------------------------------
// Phase 2: single block reduces the 8192 logits to the scalar loss.
// ---------------------------------------------------------------------------
__global__ void __launch_bounds__(256)
clip_finalize_kernel(const float* __restrict__ Sv, const float* __restrict__ rs,
                     const float* __restrict__ cs, float* __restrict__ out) {
    const int tid  = threadIdx.x;
    const int lane = tid & 31;
    const int wave = tid >> 5;

    float acc = 0.0f;
    for (int i = tid; i < B_ROWS; i += 256)
        acc += 2.0f * Sv[i] - logf(rs[i]) - logf(cs[i]);

#pragma unroll
    for (int off = 16; off >= 1; off >>= 1) acc += __shfl_xor(acc, off, 32);

    __shared__ float part[8];
    if (lane == 0) part[wave] = acc;
    __syncthreads();
    if (tid == 0) {
        float t = 0.0f;
#pragma unroll
        for (int i = 0; i < 8; ++i) t += part[i];
        out[0] = -t / (2.0f * (float)B_ROWS);
    }
}

// ---------------------------------------------------------------------------
extern "C" void kernel_launch(void* const* d_in, const int* in_sizes, int n_in,
                              void* d_out, int out_size, void* d_ws, size_t ws_size,
                              hipStream_t stream) {
    (void)in_sizes; (void)n_in; (void)out_size; (void)ws_size;
    const float* X = (const float*)d_in[0];
    const float* Y = (const float*)d_in[1];
    float* out = (float*)d_out;

    char* ws = (char*)d_ws;
    const size_t FEAT_BYTES = (size_t)B_ROWS * D_DIM * sizeof(__bf16);  // 8 MB
    __bf16* Xnb = (__bf16*)(ws);
    __bf16* Ynb = (__bf16*)(ws + FEAT_BYTES);
    float*  Sv  = (float*)(ws + 2 * FEAT_BYTES);
    float*  rs  = Sv + B_ROWS;
    float*  cs  = rs + B_ROWS;

    clip_normalize_kernel<<<B_ROWS, 128, 0, stream>>>(X, Y, Xnb, Ynb, Sv, rs, cs);

    dim3 grid(B_ROWS / 128, B_ROWS / 128);   // 64 x 64 tiles of 128x128
    clip_tile_kernel<<<grid, 256, LDS_BYTES, stream>>>(Xnb, Ynb, rs, cs);

    clip_finalize_kernel<<<1, 256, 0, stream>>>(Sv, rs, cs, out);
}
